// SPDMDictConstraint_18717467476430
// MI455X (gfx1250) — matrix operand order repaired and checked by reading, not compiled, not run on hardware
//
#include <hip/hip_runtime.h>
#include <math.h>

// ---------------------------------------------------------------------------
// Problem constants (from reference setup_inputs): B=4, T=2048, D=1024,
// DICT=16384, FAM=16, top_k=8.  NT = B*T = 8192 tokens.
// Outputs (flat, in order): recon_loss[1], sparsity_loss[1],
// sparse_coeffs[NT*DICT], probs[NT*FAM].
// ---------------------------------------------------------------------------
#define NT    8192
#define DIM   1024
#define DICT  16384
#define FAM   16
#define TOPK  8

typedef unsigned short u16;
typedef __attribute__((ext_vector_type(8)))  u16    u16x8;
typedef __attribute__((ext_vector_type(16))) u16    u16x16;
typedef __attribute__((ext_vector_type(16))) __bf16 v16bf;
typedef __attribute__((ext_vector_type(8)))  float  v8f;

// round-to-nearest-even f32 -> bf16 (raw u16 storage)
__device__ __forceinline__ u16 f2bf(float f) {
    unsigned u = __builtin_bit_cast(unsigned, f);
    unsigned r = u + 0x7FFFu + ((u >> 16) & 1u);
    return (u16)(r >> 16);
}

__device__ __forceinline__ float gelu_exact(float x) {
    return 0.5f * x * (1.0f + erff(x * 0.70710678118654752f));
}

// ---- CDNA5 async global->LDS DMA (ISA 15.18.3 op 98, ASYNCcnt-tracked) ----
// vdst = VGPR holding LDS byte offset, vaddr = 64-bit global address, no SADDR.
__device__ __forceinline__ void async_load_b128(unsigned lds_off, const void* gptr) {
    asm volatile("global_load_async_to_lds_b128 %0, %1, off"
                 :: "v"(lds_off), "v"((unsigned long long)(size_t)gptr)
                 : "memory");
}
__device__ __forceinline__ void wait_async0() {
    asm volatile("s_wait_asynccnt 0" ::: "memory");
}
// Generic shared pointer -> LDS byte offset (flat LDS aperture keeps the
// offset in addr[31:0], identical to an addrspace(3) ptrtoint).
__device__ __forceinline__ unsigned lds_off_of(const void* p) {
    return (unsigned)(size_t)p;
}

// ---- WMMA fragment loaders from LDS (bf16, 16x16x32) ----------------------
// A (16xK=32, M x K): lane half h (lane>>4): elems 0..7 = K[h*8 .. h*8+7],
//                     elems 8..15 = K[h*8+16 .. h*8+23]   (ISA 7.12.2 table)
__device__ __forceinline__ v16bf load_a_frag(const u16* rowbase, int half) {
    u16x8 lo = *(const u16x8*)(rowbase + half * 8);
    u16x8 hi = *(const u16x8*)(rowbase + half * 8 + 16);
    union { u16x16 s; v16bf b; u16x8 h[2]; } u;
    u.h[0] = lo; u.h[1] = hi;
    return u.b;
}
// B (K=32 x 16): lane n holds B[k, n]; rows of B^T are contiguous K values:
//                elems 0..7 = K[half*8 .. half*8+7], elems 8..15 = K[half*8+16 .. half*8+23]
__device__ __forceinline__ v16bf load_b_frag(const u16* rowbase, int half) {
    u16x8 lo = *(const u16x8*)(rowbase + half * 8);
    u16x8 hi = *(const u16x8*)(rowbase + half * 8 + 16);
    union { u16x16 s; v16bf b; u16x8 h[2]; } u;
    u.h[0] = lo; u.h[1] = hi;
    return u.b;
}

// ---------------------------------------------------------------------------
// f32 -> bf16 conversion (grid-stride)
// ---------------------------------------------------------------------------
__global__ __launch_bounds__(256) void k_f2bf(const float* __restrict__ src,
                                              u16* __restrict__ dst, int n) {
    for (int i = blockIdx.x * 256 + threadIdx.x; i < n; i += gridDim.x * 256)
        dst[i] = f2bf(src[i]);
}

// ---------------------------------------------------------------------------
// scores = hs . family_keys^T (F=16), softmax -> probs   (one block per token)
// ---------------------------------------------------------------------------
__global__ __launch_bounds__(128) void k_scores(const float* __restrict__ hs,
                                                const float* __restrict__ fk,
                                                float* __restrict__ probs) {
    const int token = blockIdx.x;
    const int tid   = threadIdx.x;
    const float* x  = hs + (size_t)token * DIM;

    float acc[FAM];
#pragma unroll
    for (int f = 0; f < FAM; ++f) acc[f] = 0.f;

    for (int d = tid; d < DIM; d += 128) {
        float v = x[d];
#pragma unroll
        for (int f = 0; f < FAM; ++f) acc[f] += v * fk[f * DIM + d];
    }

    __shared__ float red[128];
    __shared__ float sc[FAM];
    for (int f = 0; f < FAM; ++f) {
        red[tid] = acc[f];
        __syncthreads();
        for (int s = 64; s > 0; s >>= 1) {
            if (tid < s) red[tid] += red[tid + s];
            __syncthreads();
        }
        if (tid == 0) sc[f] = red[0];
        __syncthreads();
    }

    if (tid == 0) {
        float m = sc[0];
#pragma unroll
        for (int f = 1; f < FAM; ++f) m = fmaxf(m, sc[f]);
        float e[FAM], sum = 0.f;
#pragma unroll
        for (int f = 0; f < FAM; ++f) { e[f] = expf(sc[f] - m); sum += e[f]; }
        float inv = 1.0f / sum;
#pragma unroll
        for (int f = 0; f < FAM; ++f) probs[(size_t)token * FAM + f] = e[f] * inv;
    }
}

// ---------------------------------------------------------------------------
// family_protos[f,d] = sum_c family_proj_w[f,c] * W[c,d]   grid = (4, 16)
// ---------------------------------------------------------------------------
__global__ __launch_bounds__(256) void k_protos(const float* __restrict__ proj,
                                                const float* __restrict__ W,
                                                float* __restrict__ protos) {
    const int f = blockIdx.y;
    const int d = blockIdx.x * 256 + threadIdx.x;
    const float* pr = proj + (size_t)f * DICT;
    float acc = 0.f;
    for (int c = 0; c < DICT; ++c)
        acc += pr[c] * W[(size_t)c * DIM + d];   // pr[c] scalarizes (uniform)
    protos[f * DIM + d] = acc;
}

// ---------------------------------------------------------------------------
// prototypes = probs @ family_protos ; residual = hs - prototypes (bf16)
// one block per token
// ---------------------------------------------------------------------------
__global__ __launch_bounds__(256) void k_residual(const float* __restrict__ hs,
                                                  const float* __restrict__ probs,
                                                  const float* __restrict__ protos,
                                                  float* __restrict__ prototypes,
                                                  u16* __restrict__ residual) {
    const int token = blockIdx.x;
    __shared__ float p[FAM];
    if (threadIdx.x < FAM) p[threadIdx.x] = probs[(size_t)token * FAM + threadIdx.x];
    __syncthreads();

    for (int d = threadIdx.x; d < DIM; d += 256) {
        float proto = 0.f;
#pragma unroll
        for (int f = 0; f < FAM; ++f) proto += p[f] * protos[f * DIM + d];
        size_t o = (size_t)token * DIM + d;
        prototypes[o] = proto;
        residual[o]   = f2bf(hs[o] - proto);
    }
}

// ---------------------------------------------------------------------------
// Tiled WMMA GEMM: C[M,N] = A_bf16[M,K] @ B_bf16[N,K]^T + bias
// Block tile 128x128, BK=32, 256 threads = 8 waves in 2(M) x 4(N) grid,
// each wave computes 64x32 via 4x2 v_wmma_f32_16x16x32_bf16 tiles.
// Global->LDS staging uses CDNA5 async DMA (global_load_async_to_lds_b128):
// the next K-slab streams into the alternate LDS buffer while the current
// one feeds the WMMA pipe; s_wait_asynccnt 0 + barrier closes each step.
// ACT: 0 none, 1 exact GELU.  OUTBF: 0 -> f32 Cf, 1 -> bf16 Cb.
// ---------------------------------------------------------------------------
template <int ACT, int OUTBF>
__global__ __launch_bounds__(256) void k_gemm_bf16(const u16* __restrict__ A,
                                                   const u16* __restrict__ Bm,
                                                   const float* __restrict__ bias,
                                                   float* __restrict__ Cf,
                                                   u16* __restrict__ Cb,
                                                   int N, int K) {
    __shared__ __align__(16) u16 As[2][128 * 32];
    __shared__ __align__(16) u16 Bs[2][128 * 32];

    const int tid   = threadIdx.x;
    const int wave  = tid >> 5;
    const int lane  = tid & 31;
    const int half  = lane >> 4;
    const int r     = lane & 15;
    const int waveM = wave >> 2;   // 0..1 -> 64-row halves
    const int waveN = wave & 3;    // 0..3 -> 32-col quarters

    const int rowBase = blockIdx.y * 128;
    const int colBase = blockIdx.x * 128;
    const u16* Ag = A  + (size_t)rowBase * K;
    const u16* Bg = Bm + (size_t)colBase * K;

    // each thread DMAs 2 x 16B per matrix per K-step (512 chunks / tile)
    const int ldrow0 = tid >> 2;            // 0..63
    const int ldc    = (tid & 3) << 3;      // 0,8,16,24 (bf16 elems)

    auto issue_tiles = [&](int kt, int buf) {
        const int k0 = kt * 32;
#pragma unroll
        for (int i = 0; i < 2; ++i) {
            const int row = ldrow0 + i * 64;
            async_load_b128(lds_off_of(&As[buf][row * 32 + ldc]),
                            Ag + (size_t)row * K + k0 + ldc);
            async_load_b128(lds_off_of(&Bs[buf][row * 32 + ldc]),
                            Bg + (size_t)row * K + k0 + ldc);
        }
    };

    const v8f vzero = {0.f, 0.f, 0.f, 0.f, 0.f, 0.f, 0.f, 0.f};
    v8f acc[4][2];
#pragma unroll
    for (int i = 0; i < 4; ++i)
#pragma unroll
        for (int j = 0; j < 2; ++j) acc[i][j] = vzero;

    issue_tiles(0, 0);
    wait_async0();
    __syncthreads();

    const int nk = K >> 5;
    for (int kt = 0; kt < nk; ++kt) {
        const int cur = kt & 1;
        if (kt + 1 < nk) issue_tiles(kt + 1, cur ^ 1);   // DMA overlaps WMMA

        v16bf af[4], bfr[2];
#pragma unroll
        for (int i = 0; i < 4; ++i)
            af[i] = load_a_frag(&As[cur][(waveM * 64 + i * 16 + r) * 32], half);
#pragma unroll
        for (int j = 0; j < 2; ++j)
            bfr[j] = load_b_frag(&Bs[cur][(waveN * 32 + j * 16 + r) * 32], half);

#pragma unroll
        for (int i = 0; i < 4; ++i)
#pragma unroll
            for (int j = 0; j < 2; ++j)
                acc[i][j] = __builtin_amdgcn_wmma_f32_16x16x32_bf16(
                    false, af[i], false, bfr[j], (short)0, acc[i][j], false, false);

        wait_async0();       // own DMA landed in LDS
        __syncthreads();     // everyone's DMA landed / reads of cur finished
    }

    // Epilogue: C layout — v8f elem e: row = tileM + half*8 + e, col = tileN + r
#pragma unroll
    for (int i = 0; i < 4; ++i) {
        const int rowT = rowBase + waveM * 64 + i * 16 + half * 8;
#pragma unroll
        for (int j = 0; j < 2; ++j) {
            const int col = colBase + waveN * 32 + j * 16 + r;
            const float bb = bias[col];
            v8f a = acc[i][j];
#pragma unroll
            for (int e = 0; e < 8; ++e) {
                float x = a[e] + bb;
                if (ACT == 1) x = gelu_exact(x);
                const size_t o = (size_t)(rowT + e) * N + col;
                if (OUTBF) Cb[o] = f2bf(x);
                else       Cf[o] = x;
            }
        }
    }
}

// ---------------------------------------------------------------------------
// In-place top-k (k=8) per token over DICT coeffs: find top-8, zero row,
// scatter back; save (idx,val) and per-token |val| sum.  1 block / token.
// ---------------------------------------------------------------------------
__global__ __launch_bounds__(256) void k_topk(float* __restrict__ coeffs,
                                              float* __restrict__ tkv,
                                              int* __restrict__ tki,
                                              float* __restrict__ spars_part) {
    const int token = blockIdx.x;
    const int tid   = threadIdx.x;
    float* row = coeffs + (size_t)token * DICT;

    float v[TOPK];
    int   ix[TOPK];
#pragma unroll
    for (int j = 0; j < TOPK; ++j) { v[j] = -3.4e38f; ix[j] = 0; }

    for (int i = tid; i < DICT; i += 256) {     // 64 candidates/thread
        float x = row[i];
        if (x > v[TOPK - 1]) {
            int p = TOPK - 1;
            while (p > 0 && v[p - 1] < x) { v[p] = v[p - 1]; ix[p] = ix[p - 1]; --p; }
            v[p] = x; ix[p] = i;
        }
    }

    __shared__ float sv[256 * TOPK];
    __shared__ int   si[256 * TOPK];
#pragma unroll
    for (int j = 0; j < TOPK; ++j) { sv[tid * TOPK + j] = v[j]; si[tid * TOPK + j] = ix[j]; }
    __syncthreads();

    // pairwise merge of sorted-descending 8-lists
    for (int stride = 128; stride >= 1; stride >>= 1) {
        if (tid < stride) {
            const int a0 = tid * TOPK, b0 = (tid + stride) * TOPK;
            int ai = 0, bi = 0;
            float ov[TOPK]; int oi[TOPK];
#pragma unroll
            for (int k = 0; k < TOPK; ++k) {
                float av = sv[a0 + ai], bv = sv[b0 + bi];
                if (av >= bv) { ov[k] = av; oi[k] = si[a0 + ai]; ++ai; }
                else          { ov[k] = bv; oi[k] = si[b0 + bi]; ++bi; }
            }
#pragma unroll
            for (int k = 0; k < TOPK; ++k) { sv[a0 + k] = ov[k]; si[a0 + k] = oi[k]; }
        }
        __syncthreads();
    }

    // zero the whole row, then scatter the 8 survivors
    for (int i = tid; i < DICT; i += 256) row[i] = 0.0f;
    __threadfence();
    __syncthreads();

    if (tid < TOPK) {
        float val = sv[tid];
        int   idx = si[tid];
        row[idx] = val;
        tkv[token * TOPK + tid] = val;
        tki[token * TOPK + tid] = idx;
    }
    if (tid == 0) {
        float s = 0.f;
#pragma unroll
        for (int j = 0; j < TOPK; ++j) s += fabsf(sv[j]);
        spars_part[token] = s;
    }
}

// ---------------------------------------------------------------------------
// reconstruction: offset = sum_j val_j * W[idx_j,:]; err vs hidden; partials
// ---------------------------------------------------------------------------
__global__ __launch_bounds__(256) void k_recon(const float* __restrict__ hs,
                                               const float* __restrict__ prototypes,
                                               const float* __restrict__ W,
                                               const float* __restrict__ tkv,
                                               const int* __restrict__ tki,
                                               float* __restrict__ loss_part) {
    const int token = blockIdx.x;
    const int tid   = threadIdx.x;
    __shared__ float vv[TOPK];
    __shared__ int   vi[TOPK];
    if (tid < TOPK) { vv[tid] = tkv[token * TOPK + tid]; vi[tid] = tki[token * TOPK + tid]; }
    __syncthreads();

    float errsum = 0.f;
    for (int d = tid; d < DIM; d += 256) {
        float off = 0.f;
#pragma unroll
        for (int j = 0; j < TOPK; ++j) off += vv[j] * W[(size_t)vi[j] * DIM + d];
        const size_t o = (size_t)token * DIM + d;
        float e = prototypes[o] + off - hs[o];
        errsum += e * e;
    }

    __shared__ float red[256];
    red[tid] = errsum;
    __syncthreads();
    for (int s = 128; s > 0; s >>= 1) {
        if (tid < s) red[tid] += red[tid + s];
        __syncthreads();
    }
    if (tid == 0) loss_part[token] = red[0];
}

// ---------------------------------------------------------------------------
// final deterministic reduction of the two losses (single block)
// ---------------------------------------------------------------------------
__global__ __launch_bounds__(256) void k_reduce_losses(const float* __restrict__ lp,
                                                       const float* __restrict__ sp,
                                                       float* __restrict__ out) {
    const int tid = threadIdx.x;
    float a = 0.f, b = 0.f;
    for (int i = tid; i < NT; i += 256) { a += lp[i]; b += sp[i]; }
    __shared__ float ra[256], rb[256];
    ra[tid] = a; rb[tid] = b;
    __syncthreads();
    for (int s = 128; s > 0; s >>= 1) {
        if (tid < s) { ra[tid] += ra[tid + s]; rb[tid] += rb[tid + s]; }
        __syncthreads();
    }
    if (tid == 0) {
        out[0] = ra[0] * (1.0f / ((float)NT * (float)DIM));
        out[1] = rb[0] * (1.0f / ((float)NT * (float)DICT));
    }
}

// ---------------------------------------------------------------------------
extern "C" void kernel_launch(void* const* d_in, const int* in_sizes, int n_in,
                              void* d_out, int out_size, void* d_ws, size_t ws_size,
                              hipStream_t stream) {
    (void)in_sizes; (void)n_in; (void)out_size; (void)ws_size;

    const float* hs    = (const float*)d_in[0];   // [NT, DIM]
    const float* W     = (const float*)d_in[1];   // [DICT, DIM]
    const float* fk    = (const float*)d_in[2];   // [FAM, DIM]
    const float* fproj = (const float*)d_in[3];   // [FAM, DICT]
    const float* e1w   = (const float*)d_in[4];   // [DIM, DIM]
    const float* e1b   = (const float*)d_in[5];   // [DIM]
    const float* e2w   = (const float*)d_in[6];   // [DICT, DIM]
    const float* e2b   = (const float*)d_in[7];   // [DICT]
    // d_in[8] = top_k (device scalar) -> compile-time TOPK=8

    float* out       = (float*)d_out;
    float* out_sc    = out + 2;                               // [NT, DICT]
    float* out_probs = out + 2 + (size_t)NT * DICT;           // [NT, FAM]

    // ---- workspace carve-up (all 256B-aligned by construction) ----
    char* ws = (char*)d_ws;
    float* protos     = (float*)ws;  ws += (size_t)FAM * DIM * sizeof(float);
    float* prototypes = (float*)ws;  ws += (size_t)NT * DIM * sizeof(float);
    u16*   residual   = (u16*)ws;    ws += (size_t)NT * DIM * sizeof(u16);
    u16*   hbuf       = (u16*)ws;    ws += (size_t)NT * DIM * sizeof(u16);
    u16*   e1w_bf     = (u16*)ws;    ws += (size_t)DIM * DIM * sizeof(u16);
    u16*   e2w_bf     = (u16*)ws;    ws += (size_t)DICT * DIM * sizeof(u16);
    float* tkv        = (float*)ws;  ws += (size_t)NT * TOPK * sizeof(float);
    int*   tki        = (int*)ws;    ws += (size_t)NT * TOPK * sizeof(int);
    float* loss_part  = (float*)ws;  ws += (size_t)NT * sizeof(float);
    float* spars_part = (float*)ws;  ws += (size_t)NT * sizeof(float);

    // 1) weights -> bf16
    k_f2bf<<<1024, 256, 0, stream>>>(e1w, e1w_bf, DIM * DIM);
    k_f2bf<<<4096, 256, 0, stream>>>(e2w, e2w_bf, DICT * DIM);

    // 2) family scores + softmax probs (written straight into output)
    k_scores<<<NT, 128, 0, stream>>>(hs, fk, out_probs);

    // 3) family prototypes in model space
    k_protos<<<dim3(DIM / 256, FAM), 256, 0, stream>>>(fproj, W, protos);

    // 4) prototypes per token + bf16 residual
    k_residual<<<NT, 256, 0, stream>>>(hs, out_probs, protos, prototypes, residual);

    // 5) enc1: h = GELU(residual @ enc1_w^T + b)  -> bf16    [WMMA + async DMA]
    k_gemm_bf16<1, 1><<<dim3(DIM / 128, NT / 128), 256, 0, stream>>>(
        residual, e1w_bf, e1b, nullptr, hbuf, DIM, DIM);

    // 6) enc2: coeffs = h @ enc2_w^T + b -> f32, straight into output  [WMMA + async DMA]
    k_gemm_bf16<0, 0><<<dim3(DICT / 128, NT / 128), 256, 0, stream>>>(
        hbuf, e2w_bf, e2b, out_sc, nullptr, DICT, DIM);

    // 7) in-place top-8 sparsification
    k_topk<<<NT, 256, 0, stream>>>(out_sc, tkv, tki, spars_part);

    // 8) sparse reconstruction + per-token squared error
    k_recon<<<NT, 256, 0, stream>>>(hs, prototypes, W, tkv, tki, loss_part);

    // 9) scalar losses (deterministic)
    k_reduce_losses<<<1, 256, 0, stream>>>(loss_part, spars_part, out);
}
